// SelfAttnV2_22368189677669
// MI455X (gfx1250) — matrix operand order repaired and checked by reading, not compiled, not run on hardware
//
#include <hip/hip_runtime.h>

typedef unsigned short u16;
typedef unsigned int   u32;

typedef __attribute__((ext_vector_type(16))) __bf16 bf16x16;
typedef __attribute__((ext_vector_type(8)))  float  f32x8;

__device__ __forceinline__ u16 f32_to_bf16(float f) {
  u32 u = __float_as_uint(f);
  u32 r = u + 0x7FFFu + ((u >> 16) & 1u);   // round-to-nearest-even
  return (u16)(r >> 16);
}

union FragU { struct { uint4 lo, hi; } q; bf16x16 v; };

// A-matrix 16x32 bf16 fragment (lanes 0-15: K 0..7 & 16..23; lanes 16-31: K 8..15 & 24..31)
__device__ __forceinline__ bf16x16 load_frag_a(const u16* t, int rowBase, int lane) {
  int r = rowBase + (lane & 15);
  int o = (lane >> 4) * 8;
  FragU f;
  f.q.lo = *reinterpret_cast<const uint4*>(t + r * 32 + o);
  f.q.hi = *reinterpret_cast<const uint4*>(t + r * 32 + o + 16);
  return f.v;
}
// B-matrix 32x16 bf16 fragment (lanes 0-15: K 0..7 & 16..23; lanes 16-31: K 8..15 & 24..31), N = lane&15.
// LDS tile holds B^T row-major [N][K], so this is two contiguous 16B loads.
__device__ __forceinline__ bf16x16 load_frag_b(const u16* t, int rowBase, int lane) {
  int r = rowBase + (lane & 15);
  int o = (lane >> 4) * 8;
  FragU f;
  f.q.lo = *reinterpret_cast<const uint4*>(t + r * 32 + o);
  f.q.hi = *reinterpret_cast<const uint4*>(t + r * 32 + o + 16);
  return f.v;
}

// C[M,N] = alpha * A[M,K] @ Bt[N,K]^T (+ bias[N]); A/Bt bf16 row-major, C f32 or bf16.
// Block tile 128x128x32; 8 waves (2x4), each wave 64x32 via 4x2 WMMA 16x16 tiles.
// Ping-pong LDS double buffering with register-staged global prefetch: one barrier/k-step.
template<bool OUT_BF16, bool HAS_BIAS>
__global__ __launch_bounds__(256) void gemm_bf16_nt(
    const u16* __restrict__ A, const u16* __restrict__ Bt, void* __restrict__ Cv,
    const float* __restrict__ bias,
    int lda, int ldb, int ldc, int Kd, float alpha,
    long batchA, long batchB, long batchC)
{
  __shared__ alignas(16) u16 ldsA[2][128 * 32];
  __shared__ alignas(16) u16 ldsB[2][128 * 32];

  const int tid  = threadIdx.x;
  const int lane = tid & 31;
  const int wave = tid >> 5;
  const int wm   = wave >> 2;   // 0..1 : 64-row slab
  const int wn   = wave & 3;    // 0..3 : 32-col slab

  const long z = blockIdx.z;
  A  += z * batchA;
  Bt += z * batchB;
  const long rowBase = (long)blockIdx.y * 128;
  const long colBase = (long)blockIdx.x * 128;

  // Per-thread staging: two 16B chunks per tile per matrix (512 chunks / 256 threads).
  const int r0 = tid >> 2;        // 0..63
  const int c0 = (tid & 3) * 8;   // 0,8,16,24
  const u16* gA0 = A  + (rowBase + r0) * (long)lda + c0;
  const u16* gA1 = gA0 + 64 * (long)lda;
  const u16* gB0 = Bt + (colBase + r0) * (long)ldb + c0;
  const u16* gB1 = gB0 + 64 * (long)ldb;
  const int sOff0 = r0 * 32 + c0;
  const int sOff1 = sOff0 + 64 * 32;

  f32x8 acc[4][2];
  const f32x8 zero = {0.f, 0.f, 0.f, 0.f, 0.f, 0.f, 0.f, 0.f};
#pragma unroll
  for (int mt = 0; mt < 4; ++mt)
#pragma unroll
    for (int nt = 0; nt < 2; ++nt) acc[mt][nt] = zero;

  const int ktiles = Kd >> 5;

  // Prologue: stage k-tile 0 into buffer 0.
  {
    uint4 a0 = *reinterpret_cast<const uint4*>(gA0);
    uint4 a1 = *reinterpret_cast<const uint4*>(gA1);
    uint4 b0 = *reinterpret_cast<const uint4*>(gB0);
    uint4 b1 = *reinterpret_cast<const uint4*>(gB1);
    *reinterpret_cast<uint4*>(&ldsA[0][sOff0]) = a0;
    *reinterpret_cast<uint4*>(&ldsA[0][sOff1]) = a1;
    *reinterpret_cast<uint4*>(&ldsB[0][sOff0]) = b0;
    *reinterpret_cast<uint4*>(&ldsB[0][sOff1]) = b1;
  }

  int buf = 0;
  for (int kt = 0; kt < ktiles; ++kt) {
    const bool more = (kt + 1) < ktiles;

    // Issue next tile's global loads before compute (latency overlap).
    uint4 na0, na1, nb0, nb1;
    if (more) {
      const long off = (long)(kt + 1) * 32;
      na0 = *reinterpret_cast<const uint4*>(gA0 + off);
      na1 = *reinterpret_cast<const uint4*>(gA1 + off);
      nb0 = *reinterpret_cast<const uint4*>(gB0 + off);
      nb1 = *reinterpret_cast<const uint4*>(gB1 + off);
    }
    if (kt + 2 < ktiles) {
      const long poff = (long)(kt + 2) * 32;
      __builtin_prefetch(gA0 + poff, 0, 3);   // global_prefetch_b8
      __builtin_prefetch(gA1 + poff, 0, 3);
      __builtin_prefetch(gB0 + poff, 0, 3);
      __builtin_prefetch(gB1 + poff, 0, 3);
    }

    __syncthreads();   // stores into ldsX[buf] (prev iter / prologue) are visible

    bf16x16 afrag[4], bfrag[2];
#pragma unroll
    for (int mt = 0; mt < 4; ++mt) afrag[mt] = load_frag_a(ldsA[buf], wm * 64 + mt * 16, lane);
#pragma unroll
    for (int nt = 0; nt < 2; ++nt) bfrag[nt] = load_frag_b(ldsB[buf], wn * 32 + nt * 16, lane);

#pragma unroll
    for (int mt = 0; mt < 4; ++mt)
#pragma unroll
      for (int nt = 0; nt < 2; ++nt)
        acc[mt][nt] = __builtin_amdgcn_wmma_f32_16x16x32_bf16(
            false, afrag[mt], false, bfrag[nt], (short)0, acc[mt][nt], false, false);

    // Stage next tile into the other buffer (no conflict with current readers).
    if (more) {
      const int nb = buf ^ 1;
      *reinterpret_cast<uint4*>(&ldsA[nb][sOff0]) = na0;
      *reinterpret_cast<uint4*>(&ldsA[nb][sOff1]) = na1;
      *reinterpret_cast<uint4*>(&ldsB[nb][sOff0]) = nb0;
      *reinterpret_cast<uint4*>(&ldsB[nb][sOff1]) = nb1;
    }
    buf ^= 1;
  }

  // Epilogue. C/D layout: VGPR i -> M = i (lanes 0-15) or i+8 (lanes 16-31); N = lane&15.
  const int mhalf = (lane >> 4) * 8;
  const int nlane = lane & 15;
  u16*   Cb = (u16*)Cv;
  float* Cf = (float*)Cv;
#pragma unroll
  for (int mt = 0; mt < 4; ++mt) {
#pragma unroll
    for (int nt = 0; nt < 2; ++nt) {
#pragma unroll
      for (int i = 0; i < 8; ++i) {
        const long row = rowBase + wm * 64 + mt * 16 + mhalf + i;
        const long col = colBase + wn * 32 + nt * 16 + nlane;
        float v = acc[mt][nt][i] * alpha;
        if (HAS_BIAS) v += bias[col];
        const long off = z * batchC + row * (long)ldc + col;
        if (OUT_BF16) Cb[off] = f32_to_bf16(v);
        else          Cf[off] = v;
      }
    }
  }
}

__global__ void conv_f32_bf16(const float* __restrict__ in, u16* __restrict__ out, int n) {
  int i = blockIdx.x * blockDim.x + threadIdx.x;
  if (i < n) out[i] = f32_to_bf16(in[i]);
}

// in: f32 [K][N] row-major  ->  out: bf16 [N][K] row-major
__global__ void conv_trans_f32_bf16(const float* __restrict__ in, u16* __restrict__ out,
                                    int K, int N) {
  int i = blockIdx.x * blockDim.x + threadIdx.x;
  if (i < N * K) {
    int n = i / K, k = i % K;
    out[i] = f32_to_bf16(in[(long)k * N + n]);
  }
}

// vt[b][h][s] = qkv[(b*2048+s)*3072 + 2048 + h]   (V slice, transposed per batch)
__global__ void trans_v_bf16(const u16* __restrict__ qkv, u16* __restrict__ vt) {
  int i = blockIdx.x * blockDim.x + threadIdx.x;       // over 4*1024*2048 = 2^23
  int b   = i >> 21;
  int rem = i & ((1 << 21) - 1);
  int h   = rem >> 11;
  int s   = rem & 2047;
  vt[i] = qkv[((long)(b * 2048 + s)) * 3072 + 2048 + h];
}

// One block per row of 2048 scores: max -> exp -> sum -> bf16 probs
__global__ __launch_bounds__(256) void softmax_rows(const float* __restrict__ S,
                                                    u16* __restrict__ P) {
  __shared__ float red[256];
  const long row = blockIdx.x;
  const float* s = S + row * 2048;
  u16*         p = P + row * 2048;
  const int tid = threadIdx.x;

  float v[8];
  float lmax = -3.4e38f;
#pragma unroll
  for (int j = 0; j < 8; ++j) { v[j] = s[tid + j * 256]; lmax = fmaxf(lmax, v[j]); }
  red[tid] = lmax; __syncthreads();
  for (int st = 128; st > 0; st >>= 1) {
    if (tid < st) red[tid] = fmaxf(red[tid], red[tid + st]);
    __syncthreads();
  }
  const float m = red[0]; __syncthreads();

  float lsum = 0.f;
#pragma unroll
  for (int j = 0; j < 8; ++j) { v[j] = __expf(v[j] - m); lsum += v[j]; }
  red[tid] = lsum; __syncthreads();
  for (int st = 128; st > 0; st >>= 1) {
    if (tid < st) red[tid] += red[tid + st];
    __syncthreads();
  }
  const float inv = 1.0f / red[0];
#pragma unroll
  for (int j = 0; j < 8; ++j) p[tid + j * 256] = f32_to_bf16(v[j] * inv);
}

extern "C" void kernel_launch(void* const* d_in, const int* in_sizes, int n_in,
                              void* d_out, int out_size, void* d_ws, size_t ws_size,
                              hipStream_t stream) {
  (void)in_sizes; (void)n_in; (void)out_size; (void)ws_size;

  const float* x     = (const float*)d_in[0];   // [4,2048,1024]
  const float* W_qkv = (const float*)d_in[1];   // [1024,3072]
  const float* b_qkv = (const float*)d_in[2];   // [3072]
  const float* W_out = (const float*)d_in[3];   // [1024,1024]
  const float* b_out = (const float*)d_in[4];   // [1024]
  float* out = (float*)d_out;                   // [4,2048,1024]

  char* ws = (char*)d_ws;
  // workspace layout (bytes), total 200 MB
  u16*   xb     = (u16*)  (ws + 0);              // 16 MB  x in bf16 [8192][1024]
  u16*   wqkvt  = (u16*)  (ws + 16777216L);      // 6 MB   W_qkv^T bf16 [3072][1024]
  u16*   woutt  = (u16*)  (ws + 23068672L);      // 2 MB   W_out^T bf16 [1024][1024]
  u16*   qkv    = (u16*)  (ws + 25165824L);      // 48 MB  bf16 [8192][3072]
  u16*   vt     = (u16*)  (ws + 75497472L);      // 16 MB  V^T bf16 [4][1024][2048]
  float* scores = (float*)(ws + 92274688L);      // 64 MB  f32 [4][2048][2048]
  u16*   probs  = (u16*)  (ws + 159383552L);     // 32 MB  bf16 [4][2048][2048]
  u16*   attno  = (u16*)  (ws + 192937984L);     // 16 MB  bf16 [8192][1024]

  // 1) precision conversions / weight transposes
  conv_f32_bf16<<<32768, 256, 0, stream>>>(x, xb, 8192 * 1024);
  conv_trans_f32_bf16<<<12288, 256, 0, stream>>>(W_qkv, wqkvt, 1024, 3072);
  conv_trans_f32_bf16<<<4096, 256, 0, stream>>>(W_out, woutt, 1024, 1024);

  // 2) QKV projection: [8192,3072] = xb @ W_qkv + b_qkv  (bf16 out)
  gemm_bf16_nt<true, true><<<dim3(24, 64, 1), 256, 0, stream>>>(
      xb, wqkvt, qkv, b_qkv, 1024, 1024, 3072, 1024, 1.0f, 0, 0, 0);

  // 3) transpose V slice per batch
  trans_v_bf16<<<32768, 256, 0, stream>>>(qkv, vt);

  // 4) scores = (Q @ K^T) / 32  per batch (f32 out)
  gemm_bf16_nt<false, false><<<dim3(16, 16, 4), 256, 0, stream>>>(
      qkv /*Q*/, qkv + 1024 /*K*/, scores, nullptr,
      3072, 3072, 2048, 1024, 0.03125f,
      2048L * 3072, 2048L * 3072, 2048L * 2048);

  // 5) row softmax -> bf16 probs
  softmax_rows<<<8192, 256, 0, stream>>>(scores, probs);

  // 6) attn_out = probs @ V  per batch (bf16 out)
  gemm_bf16_nt<true, false><<<dim3(8, 16, 4), 256, 0, stream>>>(
      probs, vt, attno, nullptr,
      2048, 2048, 1024, 2048, 1.0f,
      2048L * 2048, 1024L * 2048, 2048L * 1024);

  // 7) final projection: out = attn_out @ W_out + b_out (f32 out)
  gemm_bf16_nt<false, true><<<dim3(8, 64, 1), 256, 0, stream>>>(
      attno, woutt, out, b_out, 1024, 1024, 1024, 1024, 1.0f, 0, 0, 0);
}
